// FrFDConv_91001767068365
// MI455X (gfx1250) — matrix operand order repaired and checked by reading, not compiled, not run on hardware
//
#include <hip/hip_runtime.h>
#include <hip/hip_bf16.h>
#include <math.h>

typedef __attribute__((ext_vector_type(16))) __bf16        v16bf;
typedef __attribute__((ext_vector_type(8)))  float         v8f;
typedef __attribute__((ext_vector_type(4)))  unsigned int  u32x4;

union Frag {
    u32x4 q[2];
    v16bf v;
};

__device__ __forceinline__ unsigned short f2bf(float f) {
    unsigned int u = __float_as_uint(f);
    u += 0x7FFFu + ((u >> 16) & 1u);          // round-to-nearest-even
    return (unsigned short)(u >> 16);
}

// ---------------- Phase 1: FrFT(3x3) weight generation ----------------

struct cplx { float re, im; };
__device__ __forceinline__ cplx cmk(float r, float i) { cplx c; c.re = r; c.im = i; return c; }
__device__ __forceinline__ cplx cadd(cplx a, cplx b)  { return cmk(a.re + b.re, a.im + b.im); }
__device__ __forceinline__ cplx cmul(cplx a, cplx b)  { return cmk(a.re * b.re - a.im * b.im,
                                                                   a.re * b.im + a.im * b.re); }
__device__ __forceinline__ cplx cscale(cplx a, float s){ return cmk(a.re * s, a.im * s); }
__device__ __forceinline__ cplx cexpi(float t)         { return cmk(cosf(t), sinf(t)); }

// 3-point chirp FrFT: ifftshift -> *c1 -> fft3 -> *c2 -> ifft3 -> *c1*scale -> fftshift
__device__ void frft3(cplx* v, const cplx* c1, const cplx* c2, float scale) {
    cplx t0 = v[1], t1 = v[2], t2 = v[0];     // ifftshift (roll left by 1, N=3)
    cplx u0 = cmul(t0, c1[0]), u1 = cmul(t1, c1[1]), u2 = cmul(t2, c1[2]);

    const float W = 0.8660254037844386f;      // sqrt(3)/2
    const cplx wm = cmk(-0.5f, -W);           // e^{-2pi i/3}
    const cplx wp = cmk(-0.5f,  W);           // e^{+2pi i/3}

    cplx U0 = cadd(cadd(u0, u1), u2);
    cplx U1 = cadd(u0, cadd(cmul(u1, wm), cmul(u2, wp)));
    cplx U2 = cadd(u0, cadd(cmul(u1, wp), cmul(u2, wm)));

    U0 = cmul(U0, c2[0]); U1 = cmul(U1, c2[1]); U2 = cmul(U2, c2[2]);

    cplx y0 = cadd(cadd(U0, U1), U2);
    cplx y1 = cadd(U0, cadd(cmul(U1, wp), cmul(U2, wm)));
    cplx y2 = cadd(U0, cadd(cmul(U1, wm), cmul(U2, wp)));

    const float s3 = scale * (1.0f / 3.0f);   // ifft 1/N folded in
    y0 = cscale(cmul(y0, c1[0]), s3);
    y1 = cscale(cmul(y1, c1[1]), s3);
    y2 = cscale(cmul(y2, c1[2]), s3);

    v[0] = y2; v[1] = y0; v[2] = y1;          // fftshift (roll right by 1)
}

// One thread per (cout,cin) filter; writes bf16 weights in [cout][tap][ci] layout.
__global__ __launch_bounds__(256) void frft_weights(
    const float* __restrict__ spec, const float* __restrict__ alphap,
    const float* __restrict__ polarw, unsigned short* __restrict__ wbf)
{
    const int tid = blockIdx.x * 256 + threadIdx.x;
    if (tid >= 64 * 64) return;
    const int co = tid >> 6;
    const int ci = tid & 63;

    float a = alphap[0];
    a = fminf(fmaxf(a, 1.0e-4f), 2.0f - 1.0e-4f);
    const float PI = 3.14159265358979323846f;
    const float tana2 = tanf(a * PI * 0.25f);
    const float sina  = sinf(a * PI * 0.5f);

    const float n2[3] = {4.0f, 1.0f, 0.0f};   // arange(-2,1)^2
    cplx c1[3], c2[3];
    #pragma unroll
    for (int j = 0; j < 3; ++j) {
        c1[j] = cexpi(-PI * n2[j] * tana2 / 3.0f);
        c2[j] = cexpi(-PI * n2[j] / (3.0f * sina));
    }
    const float scale = 1.0f / sqrtf(fabsf(sina) + 1.0e-12f);

    cplx M[3][3];
    const float* sp = spec + (size_t)tid * 9;
    #pragma unroll
    for (int r = 0; r < 3; ++r)
        #pragma unroll
        for (int c = 0; c < 3; ++c)
            M[r][c] = cmk(sp[r * 3 + c], 0.0f);

    // along last axis (rows), then along axis -2 (columns)
    #pragma unroll
    for (int r = 0; r < 3; ++r) frft3(M[r], c1, c2, scale);
    #pragma unroll
    for (int c = 0; c < 3; ++c) {
        cplx col[3] = { M[0][c], M[1][c], M[2][c] };
        frft3(col, c1, c2, scale);
        M[0][c] = col[0]; M[1][c] = col[1]; M[2][c] = col[2];
    }

    const float pw0 = polarw[0], pw1 = polarw[1];
    #pragma unroll
    for (int r = 0; r < 3; ++r)
        #pragma unroll
        for (int c = 0; c < 3; ++c) {
            const cplx y = M[r][c];
            const float mag = sqrtf(y.re * y.re + y.im * y.im);
            const float ang = atan2f(y.im, y.re) * (1.0f / PI);
            wbf[(size_t)co * 576 + (r * 3 + c) * 64 + ci] = f2bf(mag * pw0 + ang * pw1);
        }
}

// ---------------- Phase 2: implicit-GEMM conv via bf16 WMMA ----------------

#define HW   224
#define CIN  64
#define CIP  72       // padded ci stride in LDS: 144 B = 36 banks -> conflict-min b128 reads
#define COUT 64
#define KTOT 576      // CIN * 9
#define TD   18       // 16 + halo

__global__ __launch_bounds__(256) void conv_wmma(
    const float* __restrict__ x, const unsigned short* __restrict__ wbf,
    float* __restrict__ out)
{
    __shared__ alignas(16) unsigned short sX[TD * TD * CIP];  // [y][x][ci(pad)] bf16

    const int tid = threadIdx.x;
    const int n   = blockIdx.y;
    const int tx  = blockIdx.x % 14;
    const int ty  = blockIdx.x / 14;
    const int h0  = ty * 16, w0 = tx * 16;

    // Warm L2/WGP$ with the weight array: one prefetch per 256-B line (288 lines).
    for (int i = tid; i < 288; i += 256)
        __builtin_prefetch((const char*)wbf + (size_t)i * 256, 0, 3);

    // Stage 18x18x64 halo tile to LDS as bf16, ci innermost (x' fastest over
    // the global reads for coalescing).
    for (int i = tid; i < TD * TD * CIN; i += 256) {
        const int xx = i % TD;
        const int y  = (i / TD) % TD;
        const int ci = i / (TD * TD);
        const int gh = h0 + y - 1, gw = w0 + xx - 1;
        float v = 0.0f;
        if (gh >= 0 && gh < HW && gw >= 0 && gw < HW)
            v = x[(((size_t)n * CIN + ci) * HW + gh) * HW + gw];
        sX[(y * TD + xx) * CIP + ci] = f2bf(v);
    }
    __syncthreads();

    const int lane = tid & 31;
    const int wv   = tid >> 5;          // 8 waves
    const int m0   = (wv & 3) * 16;     // cout base of this wave
    const int pix0 = (wv >> 2) * 8;     // first tile row (py) of this wave
    const int half = lane >> 4;
    const int nl   = lane & 15;

    const v8f zero = {0.f, 0.f, 0.f, 0.f, 0.f, 0.f, 0.f, 0.f};
    v8f acc[8];
    #pragma unroll
    for (int t = 0; t < 8; ++t) acc[t] = zero;

    // A-matrix row for this lane: M = m0 + nl (lanes 0-15 and 16-31 both map M=0..15)
    const unsigned short* aRow = wbf + (size_t)(m0 + nl) * KTOT;

    #pragma unroll
    for (int kc = 0; kc < 18; ++kc) {               // K = 576 in chunks of 32
        const int tap = kc >> 1;                    // filter tap 0..8
        const int ci0 = (kc & 1) << 5;              // 0 or 32
        const int dr  = tap / 3;
        const int dc  = tap % 3;

        // A frag (16x32 bf16): per-lane K pairs {half*8+0..7} and {16+half*8+0..7}
        Frag A;
        const unsigned short* ap = aRow + tap * 64 + ci0 + half * 8;
        A.q[0] = *(const u32x4*)(ap);
        A.q[1] = *(const u32x4*)(ap + 16);

        #pragma unroll
        for (int t = 0; t < 8; ++t) {
            // B frag (32x16 bf16): lane col N = nl, K = ci0 + {half*8+0..7} and {16+half*8+0..7}
            const int y  = pix0 + t + dr;
            const int xx = nl + dc;
            const unsigned short* bp = &sX[(y * TD + xx) * CIP + ci0 + half * 8];
            Frag B;
            B.q[0] = *(const u32x4*)(bp);
            B.q[1] = *(const u32x4*)(bp + 16);

            acc[t] = __builtin_amdgcn_wmma_f32_16x16x32_bf16(
                false, A.v, false, B.v, (short)0, acc[t], false, false);
        }
    }

    // C/D layout: lane -> col N = nl, VGPR r -> row M = r + 8*half.
    // Output is write-once: non-temporal stores keep L2 for x halos + weights.
    #pragma unroll
    for (int t = 0; t < 8; ++t) {
        const int py = pix0 + t;
        float* op = out + (((size_t)n * COUT + m0 + half * 8) * HW + (h0 + py)) * HW + w0 + nl;
        #pragma unroll
        for (int r = 0; r < 8; ++r)
            __builtin_nontemporal_store(acc[t][r], op + (size_t)r * HW * HW);
    }
}

// ---------------- launcher ----------------

extern "C" void kernel_launch(void* const* d_in, const int* in_sizes, int n_in,
                              void* d_out, int out_size, void* d_ws, size_t ws_size,
                              hipStream_t stream) {
    (void)in_sizes; (void)n_in; (void)out_size; (void)ws_size;
    const float* x     = (const float*)d_in[0];   // (16,64,224,224) f32
    const float* spec  = (const float*)d_in[1];   // (64,64,3,3) f32
    const float* alpha = (const float*)d_in[2];   // scalar
    const float* polar = (const float*)d_in[3];   // (1,2,1,1)
    unsigned short* wbf = (unsigned short*)d_ws;  // 64*576 bf16 = 73728 B scratch

    frft_weights<<<16, 256, 0, stream>>>(spec, alpha, polar, wbf);

    dim3 grid(14 * 14, 16);                       // 196 spatial tiles x 16 images
    conv_wmma<<<grid, 256, 0, stream>>>(x, wbf, (float*)d_out);
}
